// Decoder_34110630265159
// MI455X (gfx1250) — compile-verified, matrix operand order checked by reading, not run on hardware
//
#include <hip/hip_runtime.h>
#include <hip/hip_bf16.h>

typedef __attribute__((ext_vector_type(16))) __bf16 v16bf;
typedef __attribute__((ext_vector_type(8)))  float  v8f;
typedef int v4i_t __attribute__((vector_size(16)));   // matches async builtin param

#define N_DIM   8
#define IJ_DIM  1024
#define R_DIM   64
#define U_DIM   64
#define V_DIM   64
#define H_DIM   128
#define X_DIM   64

__device__ __forceinline__ unsigned short f2bf(float f) {
    union { __bf16 b; unsigned short u; } cv;
    cv.b = (__bf16)f;            // RNE fptrunc; compiler picks native cvt
    return cv.u;
}

__device__ __forceinline__ v8f wmma_bf16(v16bf a, v16bf b, v8f c) {
    return __builtin_amdgcn_wmma_f32_16x16x32_bf16(
        /*neg_a=*/false, a, /*neg_b=*/false, b,
        /*c_mod=*/(short)0, c, /*reuse_a=*/false, /*reuse_b=*/false);
}

// ---- CDNA5 async global->LDS staging (16 B per lane), with safe fallback ----
__device__ __forceinline__ void stage16(const void* g, void* l) {
#if __has_builtin(__builtin_amdgcn_global_load_async_to_lds_b128)
    __builtin_amdgcn_global_load_async_to_lds_b128(
        (__attribute__((address_space(1))) v4i_t*)g,
        (__attribute__((address_space(3))) v4i_t*)l,
        /*offset=*/0, /*cpol=*/0);
#else
    *(uint4*)l = *(const uint4*)g;
#endif
}

__device__ __forceinline__ void wait_async() {
#if __has_builtin(__builtin_amdgcn_global_load_async_to_lds_b128)
#if __has_builtin(__builtin_amdgcn_s_wait_asynccnt)
    __builtin_amdgcn_s_wait_asynccnt(0);
#else
    asm volatile("s_wait_asynccnt 0x0" ::: "memory");
#endif
#endif
}

// Load one 16x32 bf16 fragment (A, or B^T stored row-major-in-K) from LDS.
// ISA 7.12.2: lane L -> row (L%16); half = L/16 adds +8 to K inside each
// 16-wide K group; K pairs are contiguous, so each VGPR is one 32-bit ds load
// (compiler merges adjacent ones into ds_load_b128).
__device__ __forceinline__ v16bf load_frag(const unsigned short* base, int row,
                                           int stride, int kbase, int half) {
    union { v16bf v; unsigned int u[8]; } f;
    const unsigned short* p = base + row * stride + kbase + half * 8;
#pragma unroll
    for (int g = 0; g < 8; ++g) {
        int K = ((g & 4) << 2) + ((g & 3) << 1); // 0,2,4,6,16,18,20,22
        f.u[g] = *(const unsigned int*)(p + K);
    }
    return f.v;
}

// ---------------------------------------------------------------------------
// Prep 1: a_t[n][r][h][v] (bf16) = sum_u w1[r,u,v,h] * u_in[n,r,u,v]
// Memory bound on w1 (134 MB) -> read w1 exactly once, coalesced over h.
// ---------------------------------------------------------------------------
__global__ __launch_bounds__(128) void prep_a_kernel(
    const float* __restrict__ w1, const float* __restrict__ u_in,
    unsigned short* __restrict__ a_t) {
    int rr = blockIdx.x >> 6;        // r
    int vv = blockIdx.x & 63;        // v
    int h  = threadIdx.x;            // 0..127
    float acc[N_DIM];
#pragma unroll
    for (int n = 0; n < N_DIM; ++n) acc[n] = 0.f;
    for (int uu = 0; uu < U_DIM; ++uu) {
        float w = w1[(((rr * U_DIM + uu) * V_DIM) + vv) * H_DIM + h];
#pragma unroll
        for (int n = 0; n < N_DIM; ++n) {
            acc[n] += w * u_in[(((n * R_DIM + rr) * U_DIM + uu) * V_DIM) + vv];
        }
    }
#pragma unroll
    for (int n = 0; n < N_DIM; ++n) {
        a_t[(((n * R_DIM + rr) * H_DIM + h) * V_DIM) + vv] = f2bf(acc[n]);
    }
}

// Prep 2: v -> bf16 (same [n][ij][v] layout: K contiguous for A fragments)
__global__ __launch_bounds__(256) void prep_v_kernel(
    const float* __restrict__ v_in, unsigned short* __restrict__ vb) {
    int i = blockIdx.x * 256 + threadIdx.x;
    vb[i] = f2bf(v_in[i]);
}

// Prep 3: w2[x][r][h] -> w2b[r][x][h] bf16 (B^T layout: K=h contiguous)
__global__ __launch_bounds__(256) void prep_w2_kernel(
    const float* __restrict__ w2, unsigned short* __restrict__ w2b) {
    int d = blockIdx.x * 256 + threadIdx.x;           // dest index
    int rr = d >> 13;                                 // / (64*128)
    int x  = (d >> 7) & 63;
    int h  = d & 127;
    w2b[d] = f2bf(w2[((x * R_DIM + rr) * H_DIM) + h]);
}

// ---------------------------------------------------------------------------
// Fused main kernel: per (n, 64-row ij tile), loop r with double-buffered
// async-to-LDS staging:
//   X[64,128] = relu(Vtile[64,64] x a_t[n,r]) * r_in   (WMMA bf16)
//   OUT[64,64] += Xb[64,128] x w2b[r]                  (WMMA bf16, reg accum)
// ---------------------------------------------------------------------------
__global__ __launch_bounds__(256) void fused_kernel(
    const float* __restrict__ rin,
    const unsigned short* __restrict__ vb,
    const unsigned short* __restrict__ a_t,
    const unsigned short* __restrict__ w2b,
    float* __restrict__ out) {

    __shared__ unsigned short lds_v[64 * V_DIM];          // [m][v]         8 KB
    __shared__ float          lds_r[64 * R_DIM];          // [m][r]        16 KB
    __shared__ unsigned short lds_a[2][H_DIM * V_DIM];    // [h][v] (B^T)  32 KB
    __shared__ unsigned short lds_w2[2][X_DIM * H_DIM];   // [x][h] (B^T)  32 KB
    __shared__ unsigned short lds_x[64 * H_DIM];          // [m][h]        16 KB

    const int n   = blockIdx.x >> 4;
    const int ij0 = (blockIdx.x & 15) * 64;
    const int tid = threadIdx.x;
    const int w    = tid >> 5;
    const int lane = tid & 31;
    const int l16  = lane & 15;
    const int half = lane >> 4;

    // Async-stage a_t[n][rr] (16 KB) and w2b[rr] (16 KB) into buffer `buf`.
    auto stage_tiles = [&](int rr, int buf) {
        const unsigned short* asrc = a_t + ((n * R_DIM + rr) * H_DIM) * V_DIM;
        const unsigned short* wsrc = w2b + rr * X_DIM * H_DIM;
#pragma unroll
        for (int k = 0; k < 4; ++k) {
            int off = (tid + 256 * k) * 8;   // u16 units, 16 B chunks
            stage16(asrc + off, lds_a[buf] + off);
            stage16(wsrc + off, lds_w2[buf] + off);
        }
    };

    // Initial staging: v tile (8 KB), r tile (16 KB), and tiles for rr=0.
    {
        const unsigned short* vsrc = vb + (n * IJ_DIM + ij0) * V_DIM;
        const float*          rsrc = rin + (n * IJ_DIM + ij0) * R_DIM;
#pragma unroll
        for (int k = 0; k < 2; ++k)
            stage16(vsrc + (tid + 256 * k) * 8, lds_v + (tid + 256 * k) * 8);
#pragma unroll
        for (int k = 0; k < 4; ++k)
            stage16(rsrc + (tid + 256 * k) * 4, lds_r + (tid + 256 * k) * 4);
        stage_tiles(0, 0);
    }

    const int mb    = w >> 1;        // M tile 0..3 (GEMM1 and GEMM2)
    const int nh    = w & 1;         // h-half for GEMM1
    const int xbase = (w & 1) * 2;   // x tiles {0,1} or {2,3} for GEMM2

    v8f accO0 = {0.f, 0.f, 0.f, 0.f, 0.f, 0.f, 0.f, 0.f};
    v8f accO1 = {0.f, 0.f, 0.f, 0.f, 0.f, 0.f, 0.f, 0.f};

    bool first = true;
    v16bf A0, A1;   // GEMM1 A fragments: r-invariant once lds_v is ready

    for (int rr = 0; rr < R_DIM; ++rr) {
        const int cur = rr & 1;
        // Own async ops done -> barrier makes everyone's staging visible.
        wait_async();
        __syncthreads();

        // Kick off next iteration's tiles into the other buffer; they fly
        // during this iteration's WMMA work (tracked by ASYNCcnt).
        if (rr + 1 < R_DIM) stage_tiles(rr + 1, cur ^ 1);

        if (first) {   // hoisted after first barrier: lds_v now valid
            A0 = load_frag(lds_v, mb * 16 + l16, V_DIM, 0, half);
            A1 = load_frag(lds_v, mb * 16 + l16, V_DIM, 32, half);
            first = false;
        }

        // Per-r row scales (8 ds loads).
        float sc[8];
#pragma unroll
        for (int c = 0; c < 8; ++c)
            sc[c] = lds_r[(mb * 16 + c + 8 * half) * R_DIM + rr];

        // ---- GEMM1: X[mb, nh-half of H] = Vtile x a_t  (K = V = 64) ----
#pragma unroll
        for (int t = 0; t < 4; ++t) {
            int ht = nh * 4 + t;
            v8f acc = {0.f, 0.f, 0.f, 0.f, 0.f, 0.f, 0.f, 0.f};
            v16bf B0 = load_frag(lds_a[cur], ht * 16 + l16, V_DIM, 0, half);
            acc = wmma_bf16(A0, B0, acc);
            v16bf B1 = load_frag(lds_a[cur], ht * 16 + l16, V_DIM, 32, half);
            acc = wmma_bf16(A1, B1, acc);
            // relu, scale by r_in[n, ij, rr], convert to bf16 -> lds_x
#pragma unroll
            for (int c = 0; c < 8; ++c) {
                int m = c + 8 * half;
                float xv = fmaxf(acc[c], 0.f) * sc[c];
                lds_x[(mb * 16 + m) * H_DIM + ht * 16 + l16] = f2bf(xv);
            }
        }
        __syncthreads();   // lds_x ready for all waves

        // ---- GEMM2: OUT[mb, x tiles] += Xb x w2b  (K = H = 128) ----
        v16bf Af[4];
#pragma unroll
        for (int kk = 0; kk < 4; ++kk)
            Af[kk] = load_frag(lds_x, mb * 16 + l16, H_DIM, kk * 32, half);
#pragma unroll
        for (int kk = 0; kk < 4; ++kk) {
            v16bf Bx0 = load_frag(lds_w2[cur], (xbase + 0) * 16 + l16, H_DIM, kk * 32, half);
            accO0 = wmma_bf16(Af[kk], Bx0, accO0);
            v16bf Bx1 = load_frag(lds_w2[cur], (xbase + 1) * 16 + l16, H_DIM, kk * 32, half);
            accO1 = wmma_bf16(Af[kk], Bx1, accO1);
        }
        // Next iteration's top barrier protects lds_x and the buffer swap.
    }

    // Write out[n][ij0+m][x], f32. Each (m,x) element owned by exactly one wave.
#pragma unroll
    for (int c = 0; c < 8; ++c) {
        int m = mb * 16 + c + 8 * half;
        float* o = out + (n * IJ_DIM + ij0 + m) * X_DIM;
        o[(xbase + 0) * 16 + l16] = accO0[c];
        o[(xbase + 1) * 16 + l16] = accO1[c];
    }
}

extern "C" void kernel_launch(void* const* d_in, const int* in_sizes, int n_in,
                              void* d_out, int out_size, void* d_ws, size_t ws_size,
                              hipStream_t stream) {
    const float* rin = (const float*)d_in[0];  // [8,32,32,64]
    const float* uin = (const float*)d_in[1];  // [8,64,64,64]
    const float* vin = (const float*)d_in[2];  // [8,32,32,64]
    const float* w1  = (const float*)d_in[3];  // [64,64,64,128]
    const float* w2  = (const float*)d_in[4];  // [64,64,128]
    float* out = (float*)d_out;                // [8,32,32,64]

    // Workspace carve (bf16 as u16): a_t 8 MB, vb 1 MB, w2b 1 MB.
    unsigned short* a_t = (unsigned short*)d_ws;                       // [8][64][128][64]
    unsigned short* vb  = a_t + (size_t)N_DIM * R_DIM * H_DIM * V_DIM; // [8][1024][64]
    unsigned short* w2b = vb + (size_t)N_DIM * IJ_DIM * V_DIM;         // [64][64][128]

    prep_a_kernel<<<dim3(R_DIM * V_DIM), dim3(128), 0, stream>>>(w1, uin, a_t);
    prep_v_kernel<<<dim3((N_DIM * IJ_DIM * V_DIM) / 256), dim3(256), 0, stream>>>(vin, vb);
    prep_w2_kernel<<<dim3((R_DIM * X_DIM * H_DIM) / 256), dim3(256), 0, stream>>>(w2, w2b);
    fused_kernel<<<dim3(N_DIM * (IJ_DIM / 64)), dim3(256), 0, stream>>>(rin, vb, a_t, w2b, out);
}
